// vggcpc_56315611185208
// MI455X (gfx1250) — compile-verified, matrix operand order checked by reading, not run on hardware
//
#include <hip/hip_runtime.h>
#include <stdint.h>
#include <stddef.h>

typedef __bf16 bh;
typedef __attribute__((ext_vector_type(16))) __bf16 v16bh;
typedef __attribute__((ext_vector_type(8)))  __bf16 v8bh;
typedef __attribute__((ext_vector_type(8)))  float  v8f;
typedef __attribute__((ext_vector_type(4)))  float  v4f;
typedef __attribute__((ext_vector_type(4)))  unsigned int v4u;
typedef __attribute__((ext_vector_type(8)))  int v8i;
typedef __attribute__((ext_vector_type(4)))  int v4i;

#define BN_EPS  1e-5f
#define ALPHA_W 0.1f
#define BETA_W  0.01f

// ---------------------------------------------------------------------------
// Fragment-order LDS layout (wave32 bf16 WMMA):
//   lane L needs row (L&15), K(e) = e<8 ? g*8+e : 16+g*8+(e-8), g = L>>4.
// Tiles are stored so lane L's 16 elements are contiguous: tile[L][0..15].
// A-tile writers (lane t: r=t&15, h=t>>4) produce exactly two contiguous
// 8-element chunks:  tile[r][h*8 + 0..7]  and  tile[16+r][h*8 + 0..7].
// B tiles are pre-packed in this order in global memory (1KB per 32x16 tile).
// ---------------------------------------------------------------------------

// ---------------------------------------------------------------------------
// Implicit-GEMM 3x3 SAME conv, N=16, NHWC bf16 in (C == Cinp, zero padded),
// f32 NHWC out (+bias).  M = 16*H*W pixels, N = Cout, K = 9*Cinp.
// Weights pre-packed bf16 tiles: [coBlk][p9][cbBlk][32][16] (1KB contiguous),
// fetched by the Tensor Data Mover (wave 0). A tile gathered NHWC (aligned
// 16B vector loads, single per-lane bounds predicate).
// ---------------------------------------------------------------------------
__global__ __launch_bounds__(128)
void k_conv3x3_wmma(const bh* __restrict__ act, int H, int W,
                    const bh* __restrict__ wgt, int Cinp, int Cout,
                    const float* __restrict__ bias, float* __restrict__ out) {
  __shared__ __align__(32) bh Bs[512];        // one 32x16 tile, fragment order
  __shared__ __align__(32) bh As[4][32][16];  // per wave, fragment order
  const int Mtot = 16 * H * W;
  const int wave = threadIdx.x >> 5, lane = threadIdx.x & 31;
  const int mBase = (blockIdx.x * 4 + wave) * 16;
  const int coBase = blockIdx.y * 16;
  const int nCb = Cinp >> 5;

  const int r = lane & 15, khalf = lane >> 4;
  const int m = mBase + r;
  const int pw = m % W, ph = (m / W) % H, pn = m / (W * H);

  v8bh zz;
#pragma unroll
  for (int j = 0; j < 8; ++j) zz[j] = (bh)0.f;

  v8f acc = {0.f, 0.f, 0.f, 0.f, 0.f, 0.f, 0.f, 0.f};

  for (int p9 = 0; p9 < 9; ++p9) {
    const int kh = p9 / 3 - 1, kw = p9 % 3 - 1;
    const int hs = ph + kh, ws = pw + kw;
    const bool inb = (m < Mtot) && (hs >= 0) && (hs < H) && (ws >= 0) && (ws < W);
    const bh* arow_p = act + (((size_t)pn * H + hs) * W + ws) * Cinp + khalf * 16;

    for (int cb = 0, cbB = 0; cb < Cinp; cb += 32, ++cbB) {
      // --- B tile: TDM copies the contiguous 1KB packed tile into LDS.
      if (wave == 0) {
        const bh* wtile = wgt + (((size_t)(coBase >> 4) * 9 + p9) * nCb + cbB) * 512;
        unsigned lds = (unsigned)(size_t)&Bs[0];
        unsigned long long ga = (unsigned long long)(size_t)wtile;
        // D# group0: count=1 | lds_addr | global_addr | type=2 ("image")
        v4u g0 = {1u, lds, (unsigned)ga,
                  (unsigned)((ga >> 32) & 0x1FFFFFFu) | (2u << 30)};
        // D# group1: data_size=4B; tensor_dim0=256; tensor_dim1=1;
        //            tile_dim0=256; tile_dim1=1; tensor_dim0_stride=256
        v8i g1 = {(int)(2u << 16),
                  (int)0x01000000u,   // tensor_dim0[15:0] << 16
                  (int)0x00010000u,   // tensor_dim0[31:16]=0 | tensor_dim1[15:0]=1
                  (int)(256u << 16),  // tensor_dim1[31:16]=0 | tile_dim0=256
                  1,                  // tile_dim1=1 | tile_dim2=0
                  256,                // tensor_dim0_stride
                  0, 0};
        v4i gz = {0, 0, 0, 0};
        v8i gz8 = {0, 0, 0, 0, 0, 0, 0, 0};
        __builtin_amdgcn_tensor_load_to_lds(g0, g1, gz, gz, gz8, 0);
      }
      // --- A tile: two aligned 16B loads per lane (activation C == Cinp)
      v8bh p0 = zz, p1 = zz;
      if (inb) {
        __builtin_prefetch(arow_p + cb + 32, 0, 1);
        p0 = *(const v8bh*)(arow_p + cb);
        p1 = *(const v8bh*)(arow_p + cb + 8);
      }
      *(v8bh*)&As[wave][r][khalf * 8] = p0;
      *(v8bh*)&As[wave][16 + r][khalf * 8] = p1;

      if (wave == 0) __builtin_amdgcn_s_wait_tensorcnt(0);
      __syncthreads();

      v16bh a = *(const v16bh*)&As[wave][lane][0];
      v16bh b = *(const v16bh*)&Bs[lane * 16];
      acc = __builtin_amdgcn_wmma_f32_16x16x32_bf16(false, a, false, b, (short)0, acc,
                                                    false, false);
      __syncthreads();
    }
  }
  const int n = lane & 15, g = lane >> 4;
#pragma unroll
  for (int v = 0; v < 8; ++v) {
    int mo = mBase + g * 8 + v;
    if (mo < Mtot) out[(size_t)mo * Cout + coBase + n] = acc[v] + bias[coBase + n];
  }
}

// ---------------------------------------------------------------------------
// Head linears: out[M,Np] = A[M,K](f32) * B(bf16, packed tiles) + bias.
// REQUIRES: M multiple of 16, K multiple of 32 (true for all call sites).
// B layout: [nBlk][kBlk][32][16] fragment order; tiles streamed to LDS with
// global_load_async_to_lds_b128 (ASYNCcnt). act: 0=none 1=relu 2=sigmoid.
// ---------------------------------------------------------------------------
__global__ __launch_bounds__(32)
void k_gemm_wmma(const float* __restrict__ A, int K,
                 const bh* __restrict__ B, int Np, int Nvalid,
                 const float* __restrict__ bias, float* __restrict__ out, int act) {
  __shared__ __align__(32) bh As[32][16];
  __shared__ __align__(32) bh Bs[512];
  const int lane = threadIdx.x;
  const int mBase = blockIdx.x * 16, nBase = blockIdx.y * 16;
  const int kBlks = K >> 5;
  const int r = lane & 15, khalf = lane >> 4;
  v8f acc = {0.f, 0.f, 0.f, 0.f, 0.f, 0.f, 0.f, 0.f};

  for (int kb = 0, kB = 0; kb < K; kb += 32, ++kB) {
    const bh* tile = B + ((size_t)(nBase >> 4) * kBlks + kB) * 512;
    {
      const bh* src0 = tile + lane * 16;
      unsigned d0 = (unsigned)(size_t)&Bs[lane * 16];
      asm volatile("global_load_async_to_lds_b128 %0, %1, off"
                   :: "v"(d0), "v"(src0) : "memory");
      const bh* src1 = tile + lane * 16 + 8;
      unsigned d1 = d0 + 16;
      asm volatile("global_load_async_to_lds_b128 %0, %1, off"
                   :: "v"(d1), "v"(src1) : "memory");
    }
    const float* arow = A + (size_t)(mBase + r) * K + kb + khalf * 16;
    v4f f0 = *(const v4f*)(arow);
    v4f f1 = *(const v4f*)(arow + 4);
    v4f f2 = *(const v4f*)(arow + 8);
    v4f f3 = *(const v4f*)(arow + 12);
    v8bh p0, p1;
#pragma unroll
    for (int j = 0; j < 4; ++j) {
      p0[j] = (bh)f0[j]; p0[4 + j] = (bh)f1[j];
      p1[j] = (bh)f2[j]; p1[4 + j] = (bh)f3[j];
    }
    *(v8bh*)&As[r][khalf * 8] = p0;
    *(v8bh*)&As[16 + r][khalf * 8] = p1;

    asm volatile("s_wait_asynccnt 0x0" ::: "memory");
    __syncthreads();

    v16bh a = *(const v16bh*)&As[lane][0];
    v16bh b = *(const v16bh*)&Bs[lane * 16];
    acc = __builtin_amdgcn_wmma_f32_16x16x32_bf16(false, a, false, b, (short)0, acc,
                                                  false, false);
    __syncthreads();
  }
  const int n = lane & 15, g = lane >> 4;
#pragma unroll
  for (int v = 0; v < 8; ++v) {
    int mm = mBase + g * 8 + v;
    int col = nBase + n;
    if (col < Nvalid) {
      float val = acc[v] + (bias ? bias[col] : 0.f);
      if (act == 1) val = fmaxf(val, 0.f);
      else if (act == 2) val = 1.f / (1.f + __expf(-val));
      out[(size_t)mm * Np + col] = val;
    }
  }
}

// ------------------- weight packing (fragment-order tiles) -----------------
__global__ void k_conv_w_cvt(const float* __restrict__ w, int Cin, int Cinp, int Cout,
                             bh* __restrict__ o) {
  size_t i = (size_t)blockIdx.x * blockDim.x + threadIdx.x;
  int nCb = Cinp >> 5;
  size_t tot = (size_t)(Cout >> 4) * 9 * nCb * 512;
  if (i >= tot) return;
  int e = (int)(i & 15);
  int l = (int)((i >> 4) & 31);
  size_t t = i >> 9;
  int cbB = (int)(t % nCb);
  int p9 = (int)((t / nCb) % 9);
  int coB = (int)(t / ((size_t)nCb * 9));
  int g = l >> 4;
  int Kk = (e < 8) ? (g * 8 + e) : (16 + g * 8 + (e - 8));
  int ci = cbB * 32 + Kk;
  int co = coB * 16 + (l & 15);
  float v = (ci < Cin) ? w[((size_t)co * Cin + ci) * 9 + p9] : 0.f;
  o[i] = (bh)v;
}

// mode 1: permute K for NHWC flatten vs reference NCHW flatten (128ch x 7x7)
__global__ void k_lin_w_cvt(const float* __restrict__ w, int din, int dout,
                            int Kp, int Np, bh* __restrict__ o, int mode) {
  size_t i = (size_t)blockIdx.x * blockDim.x + threadIdx.x;
  int kBlks = Kp >> 5;
  size_t tot = (size_t)(Np >> 4) * kBlks * 512;
  if (i >= tot) return;
  int e = (int)(i & 15);
  int l = (int)((i >> 4) & 31);
  size_t t = i >> 9;
  int kB = (int)(t % kBlks);
  int nB = (int)(t / kBlks);
  int g = l >> 4;
  int Kk = (e < 8) ? (g * 8 + e) : (16 + g * 8 + (e - 8));
  int k = kB * 32 + Kk;
  int c = nB * 16 + (l & 15);
  float v = 0.f;
  if (k < din && c < dout) {
    int kr = k;
    if (mode == 1) { int ch = k % 128; int s = k / 128; kr = ch * 49 + s; }
    v = w[(size_t)c * din + kr];
  }
  o[i] = (bh)v;
}

// NCHW f32 -> NHWC bf16 with channel zero-padding (Csrc -> Cdst)
__global__ void k_nchw_to_nhwc_pad(const float* __restrict__ x, int Csrc, int Cdst,
                                   int H, int W, bh* __restrict__ o) {
  size_t i = (size_t)blockIdx.x * blockDim.x + threadIdx.x;
  size_t tot = (size_t)16 * Cdst * H * W;
  if (i >= tot) return;
  int c = (int)(i % Cdst);
  size_t pix = i / Cdst;
  int w = (int)(pix % W), h = (int)((pix / W) % H), n = (int)(pix / ((size_t)W * H));
  float v = (c < Csrc) ? x[(((size_t)n * Csrc + c) * H + h) * W + w] : 0.f;
  o[i] = (bh)v;
}

__global__ void k_bf16_to_f32(const bh* __restrict__ x, float* __restrict__ y, size_t n) {
  size_t i = (size_t)blockIdx.x * blockDim.x + threadIdx.x;
  if (i < n) y[i] = (float)x[i];
}

__global__ void k_copy_f32(const float* __restrict__ x, float* __restrict__ y, size_t n) {
  size_t i = (size_t)blockIdx.x * blockDim.x + threadIdx.x;
  if (i < n) y[i] = x[i];
}

// ------------------------------ BN / pool ----------------------------------
__global__ void k_bn_stats(const float* __restrict__ x, int C, long P,
                           float* __restrict__ mean, float* __restrict__ var) {
  int c = blockIdx.x, t = threadIdx.x;
  float s = 0.f, s2 = 0.f;
  for (long i = t; i < P; i += blockDim.x) {
    float v = x[(size_t)i * C + c];
    s += v; s2 += v * v;
  }
  __shared__ float sh[256], sh2[256];
  sh[t] = s; sh2[t] = s2; __syncthreads();
  for (int o = 128; o > 0; o >>= 1) {
    if (t < o) { sh[t] += sh[t + o]; sh2[t] += sh2[t + o]; }
    __syncthreads();
  }
  if (t == 0) { float m = sh[0] / (float)P; mean[c] = m; var[c] = sh2[0] / (float)P - m * m; }
}

__global__ void k_bn_relu(const float* __restrict__ x, int C, size_t tot,
                          const float* __restrict__ mean, const float* __restrict__ var,
                          const float* __restrict__ g, const float* __restrict__ b,
                          bh* __restrict__ y) {
  size_t i = (size_t)blockIdx.x * blockDim.x + threadIdx.x;
  if (i >= tot) return;
  int c = (int)(i % C);
  float v = (x[i] - mean[c]) * rsqrtf(var[c] + BN_EPS) * g[c] + b[c];
  y[i] = (bh)fmaxf(v, 0.f);
}

__global__ void k_maxpool(const bh* __restrict__ x, int C, int H, int W,
                          bh* __restrict__ y) {
  int Ho = H / 2, Wo = W / 2;
  size_t tot = (size_t)16 * Ho * Wo * C;
  size_t i = (size_t)blockIdx.x * blockDim.x + threadIdx.x;
  if (i >= tot) return;
  int c = (int)(i % C);
  size_t p = i / C;
  int wo = (int)(p % Wo), ho = (int)((p / Wo) % Ho), n = (int)(p / ((size_t)Wo * Ho));
  const bh* b0 = x + (((size_t)n * H + ho * 2) * W + wo * 2) * C + c;
  float v = fmaxf(fmaxf((float)b0[0], (float)b0[C]),
                  fmaxf((float)b0[(size_t)W * C], (float)b0[(size_t)W * C + C]));
  y[i] = (bh)v;
}

__global__ void k_bn1d(const float* __restrict__ x, int M, int F,
                       const float* __restrict__ g, const float* __restrict__ b,
                       float* __restrict__ y) {
  int f = blockIdx.x * blockDim.x + threadIdx.x;
  if (f >= F) return;
  float s = 0.f, s2 = 0.f;
  for (int i = 0; i < M; ++i) { float v = x[(size_t)i * F + f]; s += v; s2 += v * v; }
  float m = s / M, var = s2 / M - m * m;
  float sc = g[f] * rsqrtf(var + BN_EPS);
  for (int i = 0; i < M; ++i) y[(size_t)i * F + f] = (x[(size_t)i * F + f] - m) * sc + b[f];
}

__global__ void k_concat_cols(const float* __restrict__ a, const float* __restrict__ b,
                              int M, int F, float* __restrict__ y) {
  size_t i = (size_t)blockIdx.x * blockDim.x + threadIdx.x;
  size_t tot = (size_t)M * 2 * F;
  if (i >= tot) return;
  int col = (int)(i % (2 * F)); int row = (int)(i / (2 * F));
  y[i] = (col < F) ? a[(size_t)row * F + col] : b[(size_t)row * F + col - F];
}

// ------------------------------ similarity ---------------------------------
__global__ void k_rownorm(const float* __restrict__ x, int F, float* __restrict__ o) {
  int r = blockIdx.x, t = threadIdx.x;
  __shared__ float sh[256];
  float s = 0.f;
  for (int j = t; j < F; j += 256) { float v = x[(size_t)r * F + j]; s += v * v; }
  sh[t] = s; __syncthreads();
  for (int o2 = 128; o2 > 0; o2 >>= 1) { if (t < o2) sh[t] += sh[t + o2]; __syncthreads(); }
  if (t == 0) o[r] = sqrtf(sh[0]);
}

__global__ void k_expcos(const float* __restrict__ A, const float* __restrict__ B,
                         const float* __restrict__ nA, const float* __restrict__ nB,
                         int R, int F, float* __restrict__ o) {
  int idx = blockIdx.x * blockDim.x + threadIdx.x;
  if (idx >= R * R) return;
  int i = idx / R, j = idx % R;
  float d = 0.f;
  for (int k = 0; k < F; ++k) d += A[(size_t)i * F + k] * B[(size_t)j * F + k];
  o[idx] = __expf(d / fmaxf(nA[i] * nB[j], 1e-15f));
}

__global__ void k_wsim(const float* __restrict__ w1, const float* __restrict__ v2,
                       const float* __restrict__ w2, const float* __restrict__ v1,
                       const float* __restrict__ nw1, const float* __restrict__ nv2,
                       const float* __restrict__ nw2, const float* __restrict__ nv1,
                       float* __restrict__ o) {
  int idx = blockIdx.x * blockDim.x + threadIdx.x;
  if (idx >= 32 * 32) return;
  int i = idx / 32, j = idx % 32;
  float d1 = 0.f, d2 = 0.f;
  for (int k = 0; k < 512; ++k) {
    d1 += w1[(size_t)i * 512 + k] * v2[(size_t)j * 512 + k];
    d2 += w2[(size_t)i * 512 + k] * v1[(size_t)j * 512 + k];
  }
  float c1 = d1 / (nw1[i] * nv2[j]);
  float c2 = d2 / (nw2[i] * nv1[j]);
  o[idx] = 0.5f * (__expf(1.f - c1) + __expf(1.f - c2));
}

// -------------------------------- losses -----------------------------------
__global__ void k_l2(const float* __restrict__ sim, const float* __restrict__ simil,
                     float* __restrict__ out1) {
  int t = threadIdx.x;  // 32
  float col = 0.f;
  for (int i = 0; i < 32; ++i) col += sim[i * 32 + t] * simil[i * 32 + t];
  float val = logf(col / (simil[t * 32 + t] * 16.0f));
  __shared__ float sh[32];
  sh[t] = val; __syncthreads();
  if (t == 0) {
    float s = 0.f;
    for (int i = 0; i < 32; ++i) s += sh[i];
    out1[0] = ALPHA_W * (s / 32.0f);
  }
}

__global__ void k_bce_sig(const float* __restrict__ logits, const float* __restrict__ lab,
                          float* __restrict__ out0, float* __restrict__ sig) {
  int t = threadIdx.x;  // 512
  __shared__ float sh[512];
  float v = 0.f;
  if (t < 320) {
    int i = t / 20, j = t % 20;
    float z = logits[i * 32 + j];
    float y = lab[t];
    v = fmaxf(z, 0.f) - z * y + log1pf(__expf(-fabsf(z)));
    sig[t] = 1.f / (1.f + __expf(-z));
  }
  sh[t] = v; __syncthreads();
  for (int o = 256; o > 0; o >>= 1) { if (t < o) sh[t] += sh[t + o]; __syncthreads(); }
  if (t == 0) out0[0] = sh[0] / 320.f;
}

__global__ void k_dist(const float* __restrict__ lab, float* __restrict__ dist,
                       float* __restrict__ posw) {
  int idx = threadIdx.x;  // 256
  int i = idx / 16, j = idx % 16;
  float d = 0.f;
  for (int c = 0; c < 20; ++c) {
    float a = lab[i * 20 + c], b = lab[j * 20 + c];
    d += a * (1.f - b) + (1.f - a) * b;
  }
  dist[idx] = d;
  posw[idx] = 1.f - d / 20.f;
}

__global__ void k_l3cls(const float* __restrict__ S, const float* __restrict__ dist,
                        const float* __restrict__ posw, const float* __restrict__ lab,
                        float* __restrict__ cls) {
  int c = blockIdx.x, t = threadIdx.x;  // 256
  __shared__ int pm[16];
  __shared__ float negsum[16];
  __shared__ float red[256];
  __shared__ int redc[256];
  if (t < 16) pm[t] = (lab[t * 20 + c] == 1.0f) ? 1 : 0;
  __syncthreads();
  if (t < 16) {
    float s = 0.f;
    for (int j = 0; j < 16; ++j) if (!pm[j]) s += S[t * 16 + j] * dist[t * 16 + j];
    negsum[t] = s;
  }
  __syncthreads();
  int nneg = 0;
  for (int j = 0; j < 16; ++j) nneg += pm[j] ? 0 : 1;
  float term = 0.f; int cnt = 0;
  {
    int i = t / 16, j = t % 16;
    if (pm[i] && pm[j]) {
      float pd = S[t] * posw[t];
      float denom = negsum[i] + pd;
      float nf = (float)(nneg > 0 ? nneg : 1);
      term = logf(denom / (pd * nf));
      cnt = 1;
    }
  }
  red[t] = term; redc[t] = cnt; __syncthreads();
  for (int o = 128; o > 0; o >>= 1) {
    if (t < o) { red[t] += red[t + o]; redc[t] += redc[t + o]; }
    __syncthreads();
  }
  if (t == 0) {
    float v = red[0] / (float)(redc[0] > 0 ? redc[0] : 1);
    cls[c] = (redc[0] > 0 && nneg > 0) ? v : 0.f;
  }
}

__global__ void k_l3sum(const float* __restrict__ cls, float* __restrict__ out2) {
  if (threadIdx.x == 0) {
    float s = 0.f;
    for (int i = 0; i < 20; ++i) s += cls[i];
    out2[0] = BETA_W * s;
  }
}

// ===========================================================================
extern "C" void kernel_launch(void* const* d_in, const int* in_sizes, int n_in,
                              void* d_out, int out_size, void* d_ws, size_t ws_size,
                              hipStream_t stream) {
  (void)in_sizes; (void)n_in; (void)out_size; (void)ws_size;
  const float* Xs[4] = {(const float*)d_in[0], (const float*)d_in[1],
                        (const float*)d_in[2], (const float*)d_in[3]};
  const float* labels = (const float*)d_in[4];
  float* outp = (float*)d_out;

  char* wp = (char*)d_ws;
  auto bump = [&](size_t bytes) -> void* {
    void* r = wp; wp += (bytes + 255) & ~(size_t)255; return r;
  };
  auto gsz = [](size_t n, int b) { return (unsigned)((n + (size_t)b - 1) / (size_t)b); };

  // ---- workspace
  const size_t MAXACT = (size_t)16 * 224 * 224 * 32;
  float* act32 = (float*)bump(MAXACT * 4);
  bh* a16A = (bh*)bump(MAXACT * 2);
  bh* a16B = (bh*)bump(MAXACT * 2);

  const int Lci[13]   = {3, 32, 32, 32, 32, 32, 32, 32, 64, 64, 64, 128, 128};
  const int Lco[13]   = {32, 32, 32, 32, 32, 32, 32, 64, 64, 64, 128, 128, 128};
  const int Lpool[13] = {0, 1, 0, 1, 0, 0, 1, 0, 0, 1, 0, 0, 1};
  int Lcip[13]; size_t Loff[13]; size_t wtot = 0;
  for (int i = 0; i < 13; ++i) {
    Lcip[i] = ((Lci[i] + 31) / 32) * 32;
    Loff[i] = wtot;
    wtot += (size_t)9 * Lcip[i] * Lco[i];
  }
  bh* convW[2] = {(bh*)bump(wtot * 2), (bh*)bump(wtot * 2)};

  bh* h1W = (bh*)bump((size_t)6272 * 512 * 2);
  bh* h2W = (bh*)bump((size_t)6272 * 512 * 2);
  bh* e1W = (bh*)bump((size_t)512 * 512 * 2);
  bh* f1W = (bh*)bump((size_t)1024 * 512 * 2);
  bh* f2W = (bh*)bump((size_t)512 * 32 * 2);
  bh* wnW = (bh*)bump((size_t)512 * 512 * 2);

  float* fA   = (float*)bump((size_t)16 * 6272 * 4);
  float* fB   = (float*)bump((size_t)16 * 6272 * 4);
  float* tmp  = (float*)bump((size_t)32 * 512 * 4);
  float* x1v1 = (float*)bump((size_t)16 * 512 * 4);
  float* x1v2 = (float*)bump((size_t)16 * 512 * 4);
  float* x2v1 = (float*)bump((size_t)16 * 512 * 4);
  float* x2v2 = (float*)bump((size_t)16 * 512 * 4);
  float* x1e1 = (float*)bump((size_t)16 * 512 * 4);
  float* x1e2 = (float*)bump((size_t)16 * 512 * 4);
  float* x2e1 = (float*)bump((size_t)16 * 512 * 4);
  float* x2e2 = (float*)bump((size_t)16 * 512 * 4);
  float* af   = (float*)bump((size_t)16 * 1024 * 4);
  float* hdn  = (float*)bump((size_t)16 * 512 * 4);
  float* logit= (float*)bump((size_t)16 * 32 * 4);
  float* v1f  = (float*)bump((size_t)32 * 512 * 4);
  float* v2f  = (float*)bump((size_t)32 * 512 * 4);
  float* wn1  = (float*)bump((size_t)32 * 512 * 4);
  float* wn2  = (float*)bump((size_t)32 * 512 * 4);
  float* nv1  = (float*)bump(32 * 4);
  float* nv2  = (float*)bump(32 * 4);
  float* nw1  = (float*)bump(32 * 4);
  float* nw2  = (float*)bump(32 * 4);
  float* naf  = (float*)bump(16 * 4);
  float* simil= (float*)bump(32 * 32 * 4);
  float* simw = (float*)bump(32 * 32 * 4);
  float* Smat = (float*)bump(16 * 16 * 4);
  float* distm= (float*)bump(16 * 16 * 4);
  float* posw = (float*)bump(16 * 16 * 4);
  float* cls  = (float*)bump(20 * 4);
  float* meanb= (float*)bump(128 * 4);
  float* varb = (float*)bump(128 * 4);

  // ---- weight packing (deterministic, redone every call)
  for (int s = 0; s < 2; ++s) {
    int base = (s == 0) ? 5 : 57;  // params_v1.cw / params_v2.cw
    for (int li = 0; li < 13; ++li) {
      size_t tot = (size_t)9 * Lcip[li] * Lco[li];
      k_conv_w_cvt<<<gsz(tot, 256), 256, 0, stream>>>(
          (const float*)d_in[base + li], Lci[li], Lcip[li], Lco[li], convW[s] + Loff[li]);
    }
  }
  k_lin_w_cvt<<<gsz((size_t)6272 * 512, 256), 256, 0, stream>>>(
      (const float*)d_in[109], 6272, 512, 6272, 512, h1W, 1);
  k_lin_w_cvt<<<gsz((size_t)6272 * 512, 256), 256, 0, stream>>>(
      (const float*)d_in[113], 6272, 512, 6272, 512, h2W, 1);
  k_lin_w_cvt<<<gsz((size_t)512 * 512, 256), 256, 0, stream>>>(
      (const float*)d_in[117], 512, 512, 512, 512, e1W, 0);
  k_lin_w_cvt<<<gsz((size_t)1024 * 512, 256), 256, 0, stream>>>(
      (const float*)d_in[121], 1024, 512, 1024, 512, f1W, 0);
  k_lin_w_cvt<<<gsz((size_t)512 * 32, 256), 256, 0, stream>>>(
      (const float*)d_in[125], 512, 20, 512, 32, f2W, 0);
  k_lin_w_cvt<<<gsz((size_t)512 * 512, 256), 256, 0, stream>>>(
      (const float*)d_in[127], 512, 512, 512, 512, wnW, 0);

  // ---- VGG backbone pass
  auto run_vgg = [&](const float* xin, int wset, int pbase, float* feat) {
    int H = 224, W = 224;
    k_nchw_to_nhwc_pad<<<gsz((size_t)16 * 32 * 224 * 224, 256), 256, 0, stream>>>(
        xin, 3, 32, 224, 224, a16A);
    bh* cur = a16A; bh* oth = a16B;
    for (int li = 0; li < 13; ++li) {
      int Co = Lco[li], Cip = Lcip[li];
      int Mtot = 16 * H * W;
      dim3 cg((unsigned)((Mtot + 63) / 64), (unsigned)(Co / 16));
      k_conv3x3_wmma<<<cg, 128, 0, stream>>>(cur, H, W, convW[wset] + Loff[li], Cip, Co,
                                             (const float*)d_in[pbase + 13 + li], act32);
      k_bn_stats<<<Co, 256, 0, stream>>>(act32, Co, (long)Mtot, meanb, varb);
      size_t tot = (size_t)Mtot * Co;
      k_bn_relu<<<gsz(tot, 256), 256, 0, stream>>>(
          act32, Co, tot, meanb, varb,
          (const float*)d_in[pbase + 26 + li], (const float*)d_in[pbase + 39 + li], oth);
      { bh* t = cur; cur = oth; oth = t; }
      if (Lpool[li]) {
        size_t ptot = (size_t)16 * (H / 2) * (W / 2) * Co;
        k_maxpool<<<gsz(ptot, 256), 256, 0, stream>>>(cur, Co, H, W, oth);
        H /= 2; W /= 2;
        { bh* t = cur; cur = oth; oth = t; }
      }
    }
    k_bf16_to_f32<<<gsz((size_t)16 * 6272, 256), 256, 0, stream>>>(cur, feat, (size_t)16 * 6272);
  };

  auto linear = [&](const float* A, int M, int K, const bh* Bw, int Np, int Nv,
                    const float* bias, float* o, int act) {
    dim3 g((unsigned)(M / 16), (unsigned)(Np / 16));
    k_gemm_wmma<<<g, 32, 0, stream>>>(A, K, Bw, Np, Nv, bias, o, act);
  };
  auto mlp_bn = [&](const float* A, int M, int K, const bh* Bw, const float* bias,
                    const float* g_, const float* b_, float* o) {
    linear(A, M, K, Bw, 512, 512, bias, tmp, 1);
    k_bn1d<<<gsz(512, 128), 128, 0, stream>>>(tmp, M, 512, g_, b_, o);
  };

  const float *h1_b = (const float*)d_in[110], *h1_g = (const float*)d_in[111],
              *h1_bb = (const float*)d_in[112];
  const float *h2_b = (const float*)d_in[114], *h2_g = (const float*)d_in[115],
              *h2_bb = (const float*)d_in[116];
  const float *e1_b = (const float*)d_in[118], *e1_g = (const float*)d_in[119],
              *e1_bb = (const float*)d_in[120];
  const float *f1_b = (const float*)d_in[122], *f_g = (const float*)d_in[123],
              *f_bb = (const float*)d_in[124];
  const float *f2_b = (const float*)d_in[126], *wn_b = (const float*)d_in[128];

  // ---- view 1 / view 2 features
  run_vgg(Xs[0], 0, 5, fA);
  mlp_bn(fA, 16, 6272, h1W, h1_b, h1_g, h1_bb, x1v1);
  run_vgg(Xs[2], 1, 57, fB);
  mlp_bn(fB, 16, 6272, h2W, h2_b, h2_g, h2_bb, x1v2);
  mlp_bn(x1v1, 16, 512, e1W, e1_b, e1_g, e1_bb, x1e1);
  mlp_bn(x1v2, 16, 512, e1W, e1_b, e1_g, e1_bb, x1e2);

  k_concat_cols<<<gsz((size_t)16 * 1024, 256), 256, 0, stream>>>(x1v1, x1v2, 16, 512, af);
  linear(af, 16, 1024, f1W, 512, 512, f1_b, tmp, 1);
  k_bn1d<<<gsz(512, 128), 128, 0, stream>>>(tmp, 16, 512, f_g, f_bb, hdn);
  linear(hdn, 16, 512, f2W, 32, 20, f2_b, logit, 0);
  k_bce_sig<<<1, 512, 0, stream>>>(logit, labels, outp, outp + 3);

  // ---- second crops
  run_vgg(Xs[1], 0, 5, fA);
  mlp_bn(fA, 16, 6272, h1W, h1_b, h1_g, h1_bb, x2v1);
  run_vgg(Xs[3], 1, 57, fB);
  mlp_bn(fB, 16, 6272, h2W, h2_b, h2_g, h2_bb, x2v2);
  mlp_bn(x2v1, 16, 512, e1W, e1_b, e1_g, e1_bb, x2e1);
  mlp_bn(x2v2, 16, 512, e1W, e1_b, e1_g, e1_bb, x2e2);

  k_copy_f32<<<gsz((size_t)16 * 512, 256), 256, 0, stream>>>(x1e1, v1f, (size_t)16 * 512);
  k_copy_f32<<<gsz((size_t)16 * 512, 256), 256, 0, stream>>>(x2e1, v1f + 16 * 512, (size_t)16 * 512);
  k_copy_f32<<<gsz((size_t)16 * 512, 256), 256, 0, stream>>>(x1e2, v2f, (size_t)16 * 512);
  k_copy_f32<<<gsz((size_t)16 * 512, 256), 256, 0, stream>>>(x2e2, v2f + 16 * 512, (size_t)16 * 512);

  // ---- l2 (NCE with weight-net gating)
  k_rownorm<<<32, 256, 0, stream>>>(v1f, 512, nv1);
  k_rownorm<<<32, 256, 0, stream>>>(v2f, 512, nv2);
  k_expcos<<<gsz(32 * 32, 256), 256, 0, stream>>>(v1f, v2f, nv1, nv2, 32, 512, simil);
  linear(v1f, 32, 512, wnW, 512, 512, wn_b, wn1, 2);
  linear(v2f, 32, 512, wnW, 512, 512, wn_b, wn2, 2);
  k_rownorm<<<32, 256, 0, stream>>>(wn1, 512, nw1);
  k_rownorm<<<32, 256, 0, stream>>>(wn2, 512, nw2);
  k_wsim<<<gsz(32 * 32, 256), 256, 0, stream>>>(wn1, v2f, wn2, v1f, nw1, nv2, nw2, nv1, simw);
  k_l2<<<1, 32, 0, stream>>>(simw, simil, outp + 1);

  // ---- l3 (per-class Hamming-weighted contrastive)
  k_rownorm<<<16, 256, 0, stream>>>(af, 1024, naf);
  k_expcos<<<gsz(16 * 16, 256), 256, 0, stream>>>(af, af, naf, naf, 16, 1024, Smat);
  k_dist<<<1, 256, 0, stream>>>(labels, distm, posw);
  k_l3cls<<<20, 256, 0, stream>>>(Smat, distm, posw, labels, cls);
  k_l3sum<<<1, 32, 0, stream>>>(cls, outp + 2);
}
